// ResidualBlockBucket_73057393705082
// MI455X (gfx1250) — matrix operand order repaired and checked by reading, not compiled, not run on hardware
//
#include <hip/hip_runtime.h>
#include <stdint.h>

#define BB 4
#define CC 64
#define HH 192
#define WW 192
#define TT 216
#define CKK 576
#define HWN (HH*WW)          // 36864
#define NPIX (BB*HH*WW)      // 147456
#define TILE 64

typedef __attribute__((ext_vector_type(16))) __bf16          v16bf;
typedef __attribute__((ext_vector_type(8)))  float           v8f;
typedef __attribute__((ext_vector_type(8)))  unsigned short  us8;
typedef __attribute__((ext_vector_type(16))) unsigned short  us16;
typedef __attribute__((ext_vector_type(4)))  unsigned int    u32x4;
typedef __attribute__((ext_vector_type(8)))  int             i32x8;
typedef __attribute__((ext_vector_type(4)))  int             i32x4;

static __device__ __forceinline__ unsigned short f2bf(float f) {
  union { float f; unsigned u; } v; v.f = f;
  unsigned u = v.u;
  u += 0x7fffu + ((u >> 16) & 1u);       // round-to-nearest-even
  return (unsigned short)(u >> 16);
}

static __device__ __forceinline__ v8f wmma_bf16(us16 a, us16 b, v8f c) {
  return __builtin_amdgcn_wmma_f32_16x16x32_bf16(
      false, __builtin_bit_cast(v16bf, a),
      false, __builtin_bit_cast(v16bf, b),
      (short)0, c, false, false);
}

// ---- gfx1250 async global->LDS copy of 16B (ASYNCcnt path) ----
static __device__ __forceinline__ void async_copy_b128(void* lds_ptr, const void* gptr) {
  unsigned loff = (unsigned)(uintptr_t)lds_ptr;          // low 32 bits = LDS offset
  unsigned long long ga = (unsigned long long)(uintptr_t)gptr;
  asm volatile("global_load_async_to_lds_b128 %0, %1, off"
               :: "v"(loff), "v"(ga) : "memory");
}
static __device__ __forceinline__ void wait_async0() {
  asm volatile("s_wait_asynccnt 0" ::: "memory");
}

// ---- gfx1250 Tensor Data Mover: 1-D contiguous copy (8B elems) ----
static __device__ __forceinline__ void tdm_load_1d(void* lds_ptr, const void* gptr,
                                                   unsigned bytes) {
  unsigned nelem = bytes >> 3;                           // 8-byte elements, <= 65535
  unsigned loff = (unsigned)(uintptr_t)lds_ptr;
  unsigned long long ga = (unsigned long long)(uintptr_t)gptr;
  u32x4 g0;
  g0[0] = 1u;                                            // count=1 (valid user D#)
  g0[1] = loff;                                          // lds_addr
  g0[2] = (unsigned)(ga & 0xffffffffu);                  // global_addr[31:0]
  g0[3] = (unsigned)((ga >> 32) & 0x1ffffffu) | (2u << 30); // addr[56:32] | type=2
  i32x8 g1;
  g1[0] = (int)(3u << 16);                               // data_size = 8B
  g1[1] = (int)((nelem & 0xffffu) << 16);                // tensor_dim0 lo16
  g1[2] = (int)((nelem >> 16) | (1u << 16));             // dim0 hi16 | tensor_dim1=1
  g1[3] = (int)(nelem << 16);                            // tile_dim0
  g1[4] = 0;                                             // tile_dim1/2 unused
  g1[5] = (int)nelem;                                    // tensor_dim0_stride lo32
  g1[6] = 0; g1[7] = 0;
  i32x4 z4 = {0, 0, 0, 0};
#if defined(__clang_major__) && (__clang_major__ >= 23)
  i32x8 z8 = {0, 0, 0, 0, 0, 0, 0, 0};
  __builtin_amdgcn_tensor_load_to_lds(g0, g1, z4, z4, z8, 0);
#else
  __builtin_amdgcn_tensor_load_to_lds(g0, g1, z4, z4, 0);
#endif
}

// ---------------- kernel 0: NCHW f32 -> NHWC bf16 ----------------
__global__ void k_pack_nhwc_bf16(const float* __restrict__ x,
                                 unsigned short* __restrict__ dst) {
  int e = blockIdx.x * 256 + threadIdx.x;        // NHWC-ordered id
  int c = e & 63;
  int w = (e >> 6) % WW;
  int h = (e >> 6) / WW % HH;
  int b = e / (64 * WW * HH);
  dst[e] = f2bf(x[(((size_t)b*CC + c)*HH + h)*WW + w]);
}

// ---- conv weights f32 (O,I,3,3) -> bf16 tap-major [t][co][ci] ----
__global__ void k_cvt_wtap(const float* __restrict__ w, unsigned short* __restrict__ o) {
  int e = blockIdx.x * 256 + threadIdx.x;        // over 36864
  int t  = e % 9;
  int ci = (e / 9) & 63;
  int co = e / CKK;
  o[(t * 64 + co) * 64 + ci] = f2bf(w[e]);
}

// ---- emb_w f32 (T,64,ci*9+tap) -> bf16 (T,64, tap*64+ci) ----
__global__ void k_cvt_emb(const float* __restrict__ ew, unsigned short* __restrict__ o) {
  size_t e = (size_t)blockIdx.x * 256 + threadIdx.x;   // over 216*64*576
  int kp = (int)(e % CKK);                             // tap*64+ci
  int co = (int)((e / CKK) % 64);
  int t  = (int)(e / (CKK * 64));
  int tap = kp >> 6, ci = kp & 63;
  o[e] = f2bf(ew[((size_t)(t * 64 + co)) * CKK + ci * 9 + tap]);
}

// ---------------- fused conv3x3 + bias + PReLU (implicit GEMM) ----------------
__global__ void __launch_bounds__(256)
k_conv3_prelu(const unsigned short* __restrict__ src,   // NHWC bf16
              const unsigned short* __restrict__ wtap,  // bf16 [9][64][64]
              const float* __restrict__ bias,           // (64)
              const float* __restrict__ alpha_p,        // scalar
              unsigned short* __restrict__ dst) {       // NHWC bf16
  __shared__ __align__(32) unsigned short sA[9][64][64];   // tap, co, ci  (72 KB)
  __shared__ __align__(32) unsigned short sX[3][66][64];   // row, pos, ch (25 KB)

  const int tid = threadIdx.x;
  const int w0  = blockIdx.x * TILE;
  const int h   = blockIdx.y;
  const int b   = blockIdx.z;
  const us8 zz = {};

  // weights: one TDM 1-D descriptor moves the whole 72KB panel
  if (tid < 32) tdm_load_1d(&sA[0][0][0], wtap, 9 * 64 * 64 * 2);

  // input halo tile: async global->LDS rows, zero-store the SAME-pad edges
  for (int r = 0; r < 3; ++r) {
    int hh = h + r - 1;
    if (hh < 0 || hh >= HH) {
      for (int c = tid; c < (66 * 64 * 2) / 16; c += 256)
        *(us8*)((char*)&sX[r][0][0] + c * 16) = zz;
    } else {
      int lo  = (w0 == 0) ? 1 : 0;
      int hiP = (w0 + TILE >= WW) ? 64 : 65;
      if (lo == 1 && tid < 8)
        *(us8*)((char*)&sX[r][0][0] + tid * 16) = zz;
      if (hiP == 64 && tid >= 8 && tid < 16)
        *(us8*)((char*)&sX[r][65][0] + (tid - 8) * 16) = zz;
      const unsigned short* gb = src + (((size_t)b*HH + hh)*WW + (w0 - 1 + lo)) * 64;
      char* lb = (char*)&sX[r][lo][0];
      int chunks = (hiP - lo + 1) * 8;                  // 128B per position
      for (int c = tid; c < chunks; c += 256)
        async_copy_b128(lb + c * 16, (const char*)gb + c * 16);
    }
  }
  wait_async0();
  __builtin_amdgcn_s_wait_tensorcnt(0);
  __syncthreads();

  const int wave = tid >> 5, lane = tid & 31;
  const int lr = lane & 15, hi = lane >> 4;
  const int msub = wave & 3;
  const int n0 = (wave >> 2) * 2;

  v8f acc0 = {}; v8f acc1 = {};
  const int am = msub * 16 + lr;

  #pragma unroll
  for (int t = 0; t < 9; ++t) {
    const int kh = t / 3, kw = t % 3;
    #pragma unroll
    for (int s = 0; s < 2; ++s) {
      union { us16 v; us8 h[2]; } af;
      const unsigned short* ap = &sA[t][am][s*32 + hi*16];
      af.h[0] = *(const us8*)(ap);
      af.h[1] = *(const us8*)(ap + 8);
      const int chb = s*32 + hi*16;
      us16 bv0 = *(const us16*)&sX[kh][(n0*16 + lr) + kw][chb];
      us16 bv1 = *(const us16*)&sX[kh][((n0+1)*16 + lr) + kw][chb];
      acc0 = wmma_bf16(af.v, bv0, acc0);
      acc1 = wmma_bf16(af.v, bv1, acc1);
    }
  }

  const float alpha = alpha_p[0];
  const int cbase = msub*16 + hi*8;
  #pragma unroll
  for (int tl = 0; tl < 2; ++tl) {
    v8f acc = tl ? acc1 : acc0;
    int p = (n0 + tl)*16 + lr;
    us8 o;
    #pragma unroll
    for (int v = 0; v < 8; ++v) {
      float y = acc[v] + bias[cbase + v];
      y = (y >= 0.f) ? y : alpha * y;
      o[v] = f2bf(y);
    }
    *(us8*)&dst[(((size_t)b*HH + h)*WW + (w0 + p))*64 + cbase] = o;
  }
}

// ---------------- bucket sort machinery ----------------
__global__ void k_zero_counts(int* counts, int* cursors) {
  int i = threadIdx.x;
  if (i < TT) { counts[i] = 0; cursors[i] = 0; }
}
__global__ void k_hist(const int* __restrict__ bk, int* counts) {
  int i = blockIdx.x * 256 + threadIdx.x;
  if (i < NPIX) atomicAdd(&counts[bk[i]], 1);
}
__global__ void k_scan(const int* __restrict__ counts, int* __restrict__ starts) {
  if (threadIdx.x == 0) {
    int s = 0;
    for (int i = 0; i < TT; ++i) { starts[i] = s; s += counts[i]; }
    starts[TT] = s;
  }
}
__global__ void k_scatter(const int* __restrict__ bk, const int* __restrict__ starts,
                          int* cursors, int* __restrict__ sorted) {
  int i = blockIdx.x * 256 + threadIdx.x;
  if (i < NPIX) {
    int t = bk[i];
    sorted[starts[t] + atomicAdd(&cursors[t], 1)] = i;
  }
}

// ---------------- per-bucket dense GEMM (64co x 64pix x K576) + residual ----------------
__global__ void __launch_bounds__(256)
k_bucket_gemm(const unsigned short* __restrict__ act,   // conv2 output, NHWC bf16
              const unsigned short* __restrict__ embT,  // bf16 (216,64, tap*64+ci)
              const int* __restrict__ sorted,
              const int* __restrict__ starts,
              const float* __restrict__ emb_b,          // (216, 64)
              const float* __restrict__ xres,           // NCHW f32
              float* __restrict__ out) {                // NCHW f32
  __shared__ __align__(32) unsigned short sW[64][CKK];  // 72 KB
  __shared__ __align__(32) unsigned short sP[64][CKK];  // 72 KB
  __shared__ int sPid[64];

  const int t = blockIdx.x;
  const int start = starts[t];
  const int cnt = starts[t + 1] - start;
  const int p0 = blockIdx.y * 64;
  if (p0 >= cnt) return;                 // block-uniform exit

  const int tid = threadIdx.x;
  const us8 zz = {};
  if (tid < 64) sPid[tid] = (p0 + tid < cnt) ? sorted[start + p0 + tid] : -1;
  __syncthreads();

  // bucket filter panel: one TDM 1-D copy of 72KB (already bf16, A-layout)
  if (tid < 32) tdm_load_1d(&sW[0][0], embT + (size_t)t * 64 * CKK, 64 * CKK * 2);

  // gather 64 patches: 9 channel-contiguous 128B segments each, async to LDS
  for (int c = tid; c < 64 * 9 * 8; c += 256) {
    int off = c & 7;
    int seg = c >> 3;
    int tap = seg % 9;
    int p   = seg / 9;
    int pid = sPid[p];
    char* lptr = (char*)&sP[p][tap * 64] + off * 16;
    bool valid = false;
    if (pid >= 0) {
      int kh = tap / 3, kw = tap % 3;
      int bimg = pid / HWN, rem = pid % HWN;
      int hq = rem / WW, wq = rem % WW;
      int hh = hq + kh - 1, ww = wq + kw - 1;
      if (hh >= 0 && hh < HH && ww >= 0 && ww < WW) {
        async_copy_b128(lptr,
            (const char*)(act + (((size_t)bimg*HH + hh)*WW + ww) * 64) + off * 16);
        valid = true;
      }
    }
    if (!valid) *(us8*)lptr = zz;
  }
  wait_async0();
  __builtin_amdgcn_s_wait_tensorcnt(0);
  __syncthreads();

  const int wave = tid >> 5, lane = tid & 31;
  const int lr = lane & 15, hi = lane >> 4;
  const int msub = wave & 3;
  const int n0 = (wave >> 2) * 2;
  v8f acc0 = {}, acc1 = {};
  const int am = msub * 16 + lr;

  #pragma unroll 6
  for (int ks = 0; ks < 18; ++ks) {
    const int k0 = ks * 32;
    union { us16 v; us8 h[2]; } af;
    const unsigned short* ap = &sW[am][k0 + hi*16];
    af.h[0] = *(const us8*)(ap);
    af.h[1] = *(const us8*)(ap + 8);
    us16 bv0 = *(const us16*)&sP[n0*16 + lr][k0 + hi*16];
    us16 bv1 = *(const us16*)&sP[(n0+1)*16 + lr][k0 + hi*16];
    acc0 = wmma_bf16(af.v, bv0, acc0);
    acc1 = wmma_bf16(af.v, bv1, acc1);
  }

  const int cbase = msub*16 + hi*8;
  #pragma unroll
  for (int tl = 0; tl < 2; ++tl) {
    v8f acc = tl ? acc1 : acc0;
    int col = (n0 + tl)*16 + lr;
    int pid = sPid[col];
    if (pid < 0) continue;
    int bimg = pid / HWN;
    int rem  = pid % HWN;
    #pragma unroll
    for (int v = 0; v < 8; ++v) {
      int co = cbase + v;
      size_t idx = (((size_t)bimg*64 + co)*HWN) + rem;
      out[idx] = acc[v] + emb_b[t*64 + co] + xres[idx];   // RES_SCALE = 1
    }
  }
}

extern "C" void kernel_launch(void* const* d_in, const int* in_sizes, int n_in,
                              void* d_out, int out_size, void* d_ws, size_t ws_size,
                              hipStream_t stream) {
  (void)in_sizes; (void)n_in; (void)out_size; (void)ws_size;
  const float* x       = (const float*)d_in[0];
  const int*   buckets = (const int*)  d_in[1];
  const float* w1      = (const float*)d_in[2];
  const float* b1      = (const float*)d_in[3];
  const float* a1      = (const float*)d_in[4];
  const float* w2      = (const float*)d_in[5];
  const float* b2      = (const float*)d_in[6];
  const float* a2      = (const float*)d_in[7];
  const float* emb_w   = (const float*)d_in[8];
  const float* emb_b   = (const float*)d_in[9];
  float* out = (float*)d_out;

  char* ws = (char*)d_ws;
  size_t o = 0;
  unsigned short* bufA  = (unsigned short*)(ws + o); o += (size_t)NPIX * 64 * 2;
  unsigned short* bufB  = (unsigned short*)(ws + o); o += (size_t)NPIX * 64 * 2;
  unsigned short* wtap1 = (unsigned short*)(ws + o); o += 64 * CKK * 2;
  unsigned short* wtap2 = (unsigned short*)(ws + o); o += 64 * CKK * 2;
  unsigned short* embT  = (unsigned short*)(ws + o); o += (size_t)TT * 64 * CKK * 2;
  int* counts  = (int*)(ws + o); o += 1024;
  int* starts  = (int*)(ws + o); o += 1024;
  int* cursors = (int*)(ws + o); o += 1024;
  int* sorted  = (int*)(ws + o); o += (size_t)NPIX * 4;

  k_pack_nhwc_bf16<<<(NPIX * 64) / 256, 256, 0, stream>>>(x, bufA);
  k_cvt_wtap<<<(64 * CKK) / 256, 256, 0, stream>>>(w1, wtap1);
  k_cvt_wtap<<<(64 * CKK) / 256, 256, 0, stream>>>(w2, wtap2);
  k_cvt_emb<<<(TT * 64 * CKK) / 256, 256, 0, stream>>>(emb_w, embT);

  k_conv3_prelu<<<dim3(WW / TILE, HH, BB), 256, 0, stream>>>(bufA, wtap1, b1, a1, bufB);
  k_conv3_prelu<<<dim3(WW / TILE, HH, BB), 256, 0, stream>>>(bufB, wtap2, b2, a2, bufA);

  k_zero_counts<<<1, 256, 0, stream>>>(counts, cursors);
  k_hist<<<NPIX / 256, 256, 0, stream>>>(buckets, counts);
  k_scan<<<1, 32, 0, stream>>>(counts, starts);
  k_scatter<<<NPIX / 256, 256, 0, stream>>>(buckets, starts, cursors, sorted);
  // mean pixels/bucket = 683, sigma ~26 -> 16 tiles (1024 px) per bucket is ample
  k_bucket_gemm<<<dim3(TT, 16), 256, 0, stream>>>(bufA, embT, sorted, starts,
                                                  emb_b, x, out);
}
